// SDE_19344532701343
// MI455X (gfx1250) — matrix operand order repaired and checked by reading, not compiled, not run on hardware
//
#include <hip/hip_runtime.h>

// ---------------------------------------------------------------------------
// CDNA5 (gfx1250) implementation: bf16 WMMA GEMMs, async LDS B-tile fill,
// f32 atomic scatter. edge_attr kept in bf16 (traffic-bound workload).
// ---------------------------------------------------------------------------

typedef __attribute__((ext_vector_type(16))) __bf16 v16bf;
typedef __attribute__((ext_vector_type(8)))  __bf16 v8bf;
typedef __attribute__((ext_vector_type(8)))  float  v8f;
typedef int v4i_ __attribute__((vector_size(16)));  // matches builtin prototype

#define HDIM 128

__device__ __forceinline__ v8f wmma_bf16(v16bf a, v16bf b, v8f c) {
    // D = A(16x32 bf16) * B(32x16 bf16) + C(16x16 f32)
    return __builtin_amdgcn_wmma_f32_16x16x32_bf16(
        /*neg_a=*/false, a, /*neg_b=*/false, b,
        /*c_mod=*/(short)0, c, /*reuse_a=*/false, /*reuse_b=*/false);
}

// A-fragment (16x32 bf16) for one K-step from a row-major bf16 row.
// ISA layout: lane L (m = L&15, hi = L>>4): elems 0..7 -> K = 8*hi + i,
// elems 8..15 -> K = 16 + 8*hi + i (relative to kbase).
__device__ __forceinline__ v16bf load_a_bf16(const __bf16* p, int hi) {
    v8bf r0 = *(const v8bf*)(p + 8 * hi);
    v8bf r1 = *(const v8bf*)(p + 16 + 8 * hi);
    v16bf a;
#pragma unroll
    for (int i = 0; i < 8; ++i) { a[i] = r0[i]; a[i + 8] = r1[i]; }
    return a;
}

// A-fragment from fp32 row p + fp32 row q (elementwise add), converted bf16.
__device__ __forceinline__ v16bf load_a_f32x2(const float* p, const float* q, int hi) {
    v16bf a;
#pragma unroll
    for (int i = 0; i < 8; ++i) {
        float f0 = p[8 * hi + i] + q[8 * hi + i];
        float f1 = p[16 + 8 * hi + i] + q[16 + 8 * hi + i];
        a[i] = (__bf16)f0;
        a[i + 8] = (__bf16)f1;
    }
    return a;
}

// Cooperative copy of pre-swizzled B into LDS. Prefer the gfx1250 async
// global->LDS path (ASYNCcnt tracked); fall back to a VGPR round-trip copy.
__device__ __forceinline__ void copy_b_lds(__bf16* dst, const __bf16* src, int elems) {
#if __has_builtin(__builtin_amdgcn_global_load_async_to_lds_b128)
    int n = elems >> 3;  // 16B chunks
    for (int i = threadIdx.x; i < n; i += blockDim.x) {
        __builtin_amdgcn_global_load_async_to_lds_b128(
            (__attribute__((address_space(1))) v4i_*)(src + (size_t)i * 8),
            (__attribute__((address_space(3))) v4i_*)(dst + (size_t)i * 8),
            /*offset=*/0, /*cpol=*/0);
    }
#if __has_builtin(__builtin_amdgcn_s_wait_asynccnt)
    __builtin_amdgcn_s_wait_asynccnt(0);
#else
    asm volatile("s_wait_asynccnt 0" ::: "memory");
#endif
#else
    const uint4* s = (const uint4*)src;
    uint4* d = (uint4*)dst;
    int n = elems >> 3;
    for (int i = threadIdx.x; i < n; i += blockDim.x) d[i] = s[i];
#endif
}

// ---------------------------------------------------------------------------
// Small prep kernels
// ---------------------------------------------------------------------------

// Per-graph: std(t) and tval = ((sin|cos) @ t_w + t_b) @ d1_w + d1_b
__global__ void k_graph(const float* __restrict__ random_t,
                        const float* __restrict__ fourier_w,
                        const float* __restrict__ t_w, const float* __restrict__ t_b,
                        const float* __restrict__ d1_w, const float* __restrict__ d1_b,
                        float* __restrict__ stdg, float* __restrict__ tval) {
    __shared__ float fin[HDIM];
    __shared__ float red[HDIM];
    int g = blockIdx.x, j = threadIdx.x;
    float t = random_t[g] * (1.0f - 1e-5f) + 1e-5f;
    const float lg = 3.2188758248682006f;  // ln(25)
    if (j == 0) stdg[g] = sqrtf((expf(2.0f * t * lg) - 1.0f) / (2.0f * lg));
    if (j < HDIM / 2) {
        float xp = t * fourier_w[j] * 6.283185307179586f;
        fin[j] = sinf(xp);
        fin[j + HDIM / 2] = cosf(xp);
    }
    __syncthreads();
    float acc = t_b[j];
    for (int k = 0; k < HDIM; ++k) acc += fin[k] * t_w[k * HDIM + j];
    red[j] = acc * d1_w[j];
    __syncthreads();
    for (int s = HDIM / 2; s > 0; s >>= 1) {
        if (j < s) red[j] += red[j + s];
        __syncthreads();
    }
    if (j == 0) tval[g] = red[0] + d1_b[0];
}

// fp32 [K,N] weight -> bf16 WMMA-B-swizzled: flat = ((s*NT+nt)*32+lane)*16 + j
// with k = 32*s + 8*(lane>>4) + j (j<8), 32*s + 16 + 8*(lane>>4) + (j-8) (j>=8), n = nt*16 + (lane&15).
__global__ void k_swizzle(const float* __restrict__ src, __bf16* __restrict__ dst,
                          int K, int N) {
    int tid = blockIdx.x * blockDim.x + threadIdx.x;
    if (tid >= K * N) return;
    int NT = N / 16;
    int j = tid & 15;
    int lane = (tid >> 4) & 31;
    int nt = (tid >> 9) % NT;
    int s = tid / (512 * NT);
    int k = 32 * s + ((j < 8) ? (8 * (lane >> 4) + j) : (16 + 8 * (lane >> 4) + (j - 8)));
    int n = nt * 16 + (lane & 15);
    dst[tid] = (__bf16)src[k * N + n];
}

// Per-(edge,feature): distance, perturbation, rank-1 first MLP layer -> h1.
// Per-edge scalars recomputed by the 128 threads of one row (broadcast-cached
// loads); the h1 store is fully coalesced.
__global__ __launch_bounds__(256) void k_edge_prep(
    const float* __restrict__ pos, const int* __restrict__ row,
    const int* __restrict__ col, const int* __restrict__ batch,
    const float* __restrict__ z, const float* __restrict__ stdg,
    const float* __restrict__ in_w1, const float* __restrict__ in_b1,
    int* __restrict__ eg, __bf16* __restrict__ h1, int total) {
    int tid = blockIdx.x * 256 + threadIdx.x;
    if (tid >= total) return;
    int e = tid >> 7, j = tid & 127;
    int r = row[e], c = col[e];
    float dx = pos[3 * r] - pos[3 * c];
    float dy = pos[3 * r + 1] - pos[3 * c + 1];
    float dz = pos[3 * r + 2] - pos[3 * c + 2];
    float d = sqrtf(dx * dx + dy * dy + dz * dz);
    int g = batch[r];
    if (j == 0) eg[e] = g;
    float pd = d + z[e] * stdg[g];
    h1[tid] = (__bf16)fmaxf(pd * in_w1[j] + in_b1[j], 0.0f);
}

__global__ void k_gather_x(const float* __restrict__ node_emb,
                           const int* __restrict__ atom_type,
                           float* __restrict__ x, int total) {
    int tid = blockIdx.x * blockDim.x + threadIdx.x;
    if (tid >= total) return;
    int n = tid >> 7, j = tid & 127;
    x[tid] = node_emb[atom_type[n] * HDIM + j];
}

__global__ void k_zero(float* __restrict__ p, int n) {
    int tid = blockIdx.x * blockDim.x + threadIdx.x;
    if (tid < n) p[tid] = 0.0f;
}

// msg = relu(x[row] + edge_attr); agg[col] += msg  (f32 global atomics)
__global__ __launch_bounds__(256) void k_scatter(
    const float* __restrict__ x, const __bf16* __restrict__ edge_attr,
    const int* __restrict__ row, const int* __restrict__ col,
    float* __restrict__ agg, int total) {
    int tid = blockIdx.x * 256 + threadIdx.x;
    if (tid >= total) return;
    int e = tid >> 7, j = tid & 127;
    float m = x[(size_t)row[e] * HDIM + j] + (float)edge_attr[tid];
    m = fmaxf(m, 0.0f);
    unsafeAtomicAdd(&agg[(size_t)col[e] * HDIM + j], m);
}

// ---------------------------------------------------------------------------
// WMMA GEMM kernels (bf16 in, f32 accumulate). 256 threads = 8 waves;
// wave w owns rows [blk*128 + w*16, +16), computes all N-tiles; B in LDS.
// ---------------------------------------------------------------------------

// d_emb = h1 @ in_w2 + b2 + tval[g]; edge_attr = bf16(d_emb * edge_emb[etype])
__global__ __launch_bounds__(256) void k_edge_gemm(
    const __bf16* __restrict__ h1, const __bf16* __restrict__ Bsw,
    const float* __restrict__ b2, const float* __restrict__ tval,
    const int* __restrict__ eg, const int* __restrict__ etype,
    const float* __restrict__ edge_emb, __bf16* __restrict__ edge_attr, int M) {
    __shared__ __bf16 lsB[4 * 8 * 32 * 16];  // 32 KB
    copy_b_lds(lsB, Bsw, 4 * 8 * 32 * 16);
    __syncthreads();
    int wave = threadIdx.x >> 5, lane = threadIdx.x & 31;
    int m0 = blockIdx.x * 128 + wave * 16;
    int hi = lane >> 4, c0 = lane & 15;
    int arow = m0 + c0; if (arow >= M) arow = M - 1;
    v8f acc[8] = {};
#pragma unroll
    for (int s = 0; s < 4; ++s) {
        v16bf a = load_a_bf16(h1 + (size_t)arow * HDIM + 32 * s, hi);
#pragma unroll
        for (int nt = 0; nt < 8; ++nt) {
            v16bf b = *(const v16bf*)(lsB + ((s * 8 + nt) * 32 + lane) * 16);
            acc[nt] = wmma_bf16(a, b, acc[nt]);
        }
    }
#pragma unroll
    for (int v = 0; v < 8; ++v) {
        int r = m0 + hi * 8 + v;
        if (r < M) {
            float tv = tval[eg[r]];
            int et = etype[r];
#pragma unroll
            for (int nt = 0; nt < 8; ++nt) {
                int c = nt * 16 + c0;
                float dmb = acc[nt][v] + b2[c] + tv;
                edge_attr[(size_t)r * HDIM + c] = (__bf16)(dmb * edge_emb[et * HDIM + c]);
            }
        }
    }
}

// hidden = relu((x+agg) @ W1 + b1)  -> bf16
__global__ __launch_bounds__(256) void k_node_gemm1(
    const float* __restrict__ x, const float* __restrict__ agg,
    const __bf16* __restrict__ Bsw, const float* __restrict__ b1,
    __bf16* __restrict__ hidden, int M) {
    __shared__ __bf16 lsB[4 * 8 * 32 * 16];
    copy_b_lds(lsB, Bsw, 4 * 8 * 32 * 16);
    __syncthreads();
    int wave = threadIdx.x >> 5, lane = threadIdx.x & 31;
    int m0 = blockIdx.x * 128 + wave * 16;
    int hi = lane >> 4, c0 = lane & 15;
    int arow = m0 + c0; if (arow >= M) arow = M - 1;
    v8f acc[8] = {};
#pragma unroll
    for (int s = 0; s < 4; ++s) {
        v16bf a = load_a_f32x2(x + (size_t)arow * HDIM + 32 * s,
                               agg + (size_t)arow * HDIM + 32 * s, hi);
#pragma unroll
        for (int nt = 0; nt < 8; ++nt) {
            v16bf b = *(const v16bf*)(lsB + ((s * 8 + nt) * 32 + lane) * 16);
            acc[nt] = wmma_bf16(a, b, acc[nt]);
        }
    }
#pragma unroll
    for (int v = 0; v < 8; ++v) {
        int r = m0 + hi * 8 + v;
        if (r < M) {
#pragma unroll
            for (int nt = 0; nt < 8; ++nt) {
                int c = nt * 16 + c0;
                hidden[(size_t)r * HDIM + c] = (__bf16)fmaxf(acc[nt][v] + b1[c], 0.0f);
            }
        }
    }
}

// h = hidden @ W2 + b2 (optional relu); x += h  (shortcut, in-place)
__global__ __launch_bounds__(256) void k_node_gemm2(
    const __bf16* __restrict__ hidden, const __bf16* __restrict__ Bsw,
    const float* __restrict__ b2, float* __restrict__ x, int do_relu, int M) {
    __shared__ __bf16 lsB[4 * 8 * 32 * 16];
    copy_b_lds(lsB, Bsw, 4 * 8 * 32 * 16);
    __syncthreads();
    int wave = threadIdx.x >> 5, lane = threadIdx.x & 31;
    int m0 = blockIdx.x * 128 + wave * 16;
    int hi = lane >> 4, c0 = lane & 15;
    int arow = m0 + c0; if (arow >= M) arow = M - 1;
    v8f acc[8] = {};
#pragma unroll
    for (int s = 0; s < 4; ++s) {
        v16bf a = load_a_bf16(hidden + (size_t)arow * HDIM + 32 * s, hi);
#pragma unroll
        for (int nt = 0; nt < 8; ++nt) {
            v16bf b = *(const v16bf*)(lsB + ((s * 8 + nt) * 32 + lane) * 16);
            acc[nt] = wmma_bf16(a, b, acc[nt]);
        }
    }
#pragma unroll
    for (int v = 0; v < 8; ++v) {
        int r = m0 + hi * 8 + v;
        if (r < M) {
#pragma unroll
            for (int nt = 0; nt < 8; ++nt) {
                int c = nt * 16 + c0;
                float val = acc[nt][v] + b2[c];
                if (do_relu) val = fmaxf(val, 0.0f);
                size_t idx = (size_t)r * HDIM + c;
                x[idx] = x[idx] + val;
            }
        }
    }
}

// s1 = relu(concat(h_row*h_col, edge_attr) @ ow1 + ob1), K = 256
__global__ __launch_bounds__(256) void k_out_gemm1(
    const float* __restrict__ x, const int* __restrict__ row,
    const int* __restrict__ col, const __bf16* __restrict__ edge_attr,
    const __bf16* __restrict__ Bsw, const float* __restrict__ ob1,
    __bf16* __restrict__ s1, int M) {
    __shared__ __bf16 lsB[8 * 8 * 32 * 16];  // 64 KB
    copy_b_lds(lsB, Bsw, 8 * 8 * 32 * 16);
    __syncthreads();
    int wave = threadIdx.x >> 5, lane = threadIdx.x & 31;
    int m0 = blockIdx.x * 128 + wave * 16;
    int hi = lane >> 4, c0 = lane & 15;
    int e = m0 + c0; if (e >= M) e = M - 1;
    const float* xr = x + (size_t)row[e] * HDIM;
    const float* xc = x + (size_t)col[e] * HDIM;
    const __bf16* ea = edge_attr + (size_t)e * HDIM;
    v8f acc[8] = {};
#pragma unroll
    for (int s = 0; s < 8; ++s) {
        v16bf a;
        if (s < 4) {
            int kb = 32 * s;
#pragma unroll
            for (int i = 0; i < 8; ++i) {
                int k0 = kb + 8 * hi + i, k1 = kb + 16 + 8 * hi + i;
                a[i] = (__bf16)(xr[k0] * xc[k0]);
                a[i + 8] = (__bf16)(xr[k1] * xc[k1]);
            }
        } else {
            a = load_a_bf16(ea + 32 * (s - 4), hi);  // bf16 passthrough
        }
#pragma unroll
        for (int nt = 0; nt < 8; ++nt) {
            v16bf b = *(const v16bf*)(lsB + ((s * 8 + nt) * 32 + lane) * 16);
            acc[nt] = wmma_bf16(a, b, acc[nt]);
        }
    }
#pragma unroll
    for (int v = 0; v < 8; ++v) {
        int r = m0 + hi * 8 + v;
        if (r < M) {
#pragma unroll
            for (int nt = 0; nt < 8; ++nt) {
                int c = nt * 16 + c0;
                s1[(size_t)r * HDIM + c] = (__bf16)fmaxf(acc[nt][v] + ob1[c], 0.0f);
            }
        }
    }
}

// s2 = relu(s1 @ ow2 + ob2); raw = s2 @ ow3 + ob3; loss += 0.5*(raw + z)^2
// (scores*std_e == raw: std cancels algebraically). N = 64 -> 4 tiles;
// cross-lane shfl_xor reduction gives the per-row ow3 dot product.
__global__ __launch_bounds__(256) void k_out_gemm2(
    const __bf16* __restrict__ s1, const __bf16* __restrict__ Bsw,
    const float* __restrict__ ob2, const float* __restrict__ ow3,
    const float* __restrict__ ob3, const float* __restrict__ z,
    const int* __restrict__ eg, float* __restrict__ out, int M) {
    __shared__ __bf16 lsB[4 * 4 * 32 * 16];  // 16 KB
    copy_b_lds(lsB, Bsw, 4 * 4 * 32 * 16);
    __syncthreads();
    int wave = threadIdx.x >> 5, lane = threadIdx.x & 31;
    int m0 = blockIdx.x * 128 + wave * 16;
    int hi = lane >> 4, c0 = lane & 15;
    int arow = m0 + c0; if (arow >= M) arow = M - 1;
    v8f acc[4] = {};
#pragma unroll
    for (int s = 0; s < 4; ++s) {
        v16bf a = load_a_bf16(s1 + (size_t)arow * HDIM + 32 * s, hi);
#pragma unroll
        for (int nt = 0; nt < 4; ++nt) {
            v16bf b = *(const v16bf*)(lsB + ((s * 4 + nt) * 32 + lane) * 16);
            acc[nt] = wmma_bf16(a, b, acc[nt]);
        }
    }
    float w3[4], bb2[4];
#pragma unroll
    for (int nt = 0; nt < 4; ++nt) {
        w3[nt] = ow3[nt * 16 + c0];
        bb2[nt] = ob2[nt * 16 + c0];
    }
    float b3 = ob3[0];
#pragma unroll
    for (int v = 0; v < 8; ++v) {
        int r = m0 + hi * 8 + v;
        float partial = 0.0f;
#pragma unroll
        for (int nt = 0; nt < 4; ++nt) {
            float s2 = fmaxf(acc[nt][v] + bb2[nt], 0.0f);
            partial += s2 * w3[nt];
        }
#pragma unroll
        for (int off = 1; off < 16; off <<= 1)
            partial += __shfl_xor(partial, off, 32);
        if (c0 == 0 && r < M) {
            float t = (partial + b3) + z[r];
            unsafeAtomicAdd(&out[eg[r]], 0.5f * t * t);
        }
    }
}

// ---------------------------------------------------------------------------
// Host launcher
// ---------------------------------------------------------------------------
extern "C" void kernel_launch(void* const* d_in, const int* in_sizes, int n_in,
                              void* d_out, int out_size, void* d_ws, size_t ws_size,
                              hipStream_t stream) {
    const float* pos       = (const float*)d_in[0];
    const int*   atom_type = (const int*)d_in[1];
    const int*   edge_idx  = (const int*)d_in[2];
    const int*   edge_type = (const int*)d_in[3];
    const int*   batch     = (const int*)d_in[4];
    const float* random_t  = (const float*)d_in[5];
    const float* z         = (const float*)d_in[6];
    const float* node_emb  = (const float*)d_in[7];
    const float* edge_emb  = (const float*)d_in[8];
    const float* in_w1     = (const float*)d_in[9];
    const float* in_b1     = (const float*)d_in[10];
    const float* in_w2     = (const float*)d_in[11];
    const float* in_b2     = (const float*)d_in[12];
    const float* fourier_w = (const float*)d_in[13];
    const float* t_w       = (const float*)d_in[14];
    const float* t_b       = (const float*)d_in[15];
    const float* d1_w      = (const float*)d_in[16];
    const float* d1_b      = (const float*)d_in[17];
    const float* gin_w1    = (const float*)d_in[18];
    const float* gin_b1    = (const float*)d_in[19];
    const float* gin_w2    = (const float*)d_in[20];
    const float* gin_b2    = (const float*)d_in[21];
    const float* ow1       = (const float*)d_in[22];
    const float* ob1       = (const float*)d_in[23];
    const float* ow2       = (const float*)d_in[24];
    const float* ob2       = (const float*)d_in[25];
    const float* ow3       = (const float*)d_in[26];
    const float* ob3       = (const float*)d_in[27];

    const int Nn = in_sizes[1];          // 20000
    const int E  = in_sizes[3];          // 320000
    const int G  = in_sizes[5];          // 64
    const int L  = in_sizes[19] / HDIM;  // 4
    const int*   row = edge_idx;
    const int*   col = edge_idx + E;

    // Workspace carve-out (256B aligned).
    char* base = (char*)d_ws;
    size_t cur = 0;
    auto carve = [&](size_t bytes) -> void* {
        void* p = base + cur;
        cur = (cur + bytes + 255) & ~(size_t)255;
        return p;
    };
    float*  stdg      = (float*)carve((size_t)G * 4);
    float*  tval      = (float*)carve((size_t)G * 4);
    int*    eg        = (int*)carve((size_t)E * 4);
    __bf16* h1s1      = (__bf16*)carve((size_t)E * HDIM * 2);  // h1, reused as s1
    __bf16* edge_attr = (__bf16*)carve((size_t)E * HDIM * 2);
    float*  x         = (float*)carve((size_t)Nn * HDIM * 4);
    float*  agg       = (float*)carve((size_t)Nn * HDIM * 4);
    __bf16* hidden    = (__bf16*)carve((size_t)Nn * HDIM * 2);
    __bf16* w_in2     = (__bf16*)carve((size_t)HDIM * HDIM * 2);
    __bf16* w_g1      = (__bf16*)carve((size_t)L * HDIM * HDIM * 2);
    __bf16* w_g2      = (__bf16*)carve((size_t)L * HDIM * HDIM * 2);
    __bf16* w_o1      = (__bf16*)carve((size_t)2 * HDIM * HDIM * 2);
    __bf16* w_o2      = (__bf16*)carve((size_t)HDIM * (HDIM / 2) * 2);
    float*  out       = (float*)d_out;

    // 1) per-graph scalars
    k_graph<<<G, HDIM, 0, stream>>>(random_t, fourier_w, t_w, t_b, d1_w, d1_b, stdg, tval);

    // 2) weight swizzles -> bf16 WMMA-B layout
    int hh = HDIM * HDIM;
    k_swizzle<<<(hh + 255) / 256, 256, 0, stream>>>(in_w2, w_in2, HDIM, HDIM);
    for (int l = 0; l < L; ++l) {
        k_swizzle<<<(hh + 255) / 256, 256, 0, stream>>>(gin_w1 + (size_t)l * hh, w_g1 + (size_t)l * hh, HDIM, HDIM);
        k_swizzle<<<(hh + 255) / 256, 256, 0, stream>>>(gin_w2 + (size_t)l * hh, w_g2 + (size_t)l * hh, HDIM, HDIM);
    }
    k_swizzle<<<(2 * hh + 255) / 256, 256, 0, stream>>>(ow1, w_o1, 2 * HDIM, HDIM);
    k_swizzle<<<(hh / 2 + 255) / 256, 256, 0, stream>>>(ow2, w_o2, HDIM, HDIM / 2);

    // 3) per-edge prep (distance + first MLP layer), coalesced h1 stores
    int eh = E * HDIM;
    k_edge_prep<<<(eh + 255) / 256, 256, 0, stream>>>(pos, row, col, batch, z, stdg,
                                                      in_w1, in_b1, eg, h1s1, eh);

    // 4) edge GEMM -> edge_attr (bf16)
    k_edge_gemm<<<(E + 127) / 128, 256, 0, stream>>>(h1s1, w_in2, in_b2, tval, eg,
                                                     edge_type, edge_emb, edge_attr, E);

    // 5) x = node_emb[atom_type]
    int nh = Nn * HDIM;
    k_gather_x<<<(nh + 255) / 256, 256, 0, stream>>>(node_emb, atom_type, x, nh);

    // 6) GIN layers
    int ngrid = (Nn + 127) / 128;
    for (int l = 0; l < L; ++l) {
        k_zero<<<(nh + 255) / 256, 256, 0, stream>>>(agg, nh);
        k_scatter<<<(eh + 255) / 256, 256, 0, stream>>>(x, edge_attr, row, col, agg, eh);
        k_node_gemm1<<<ngrid, 256, 0, stream>>>(x, agg, w_g1 + (size_t)l * hh,
                                                gin_b1 + (size_t)l * HDIM, hidden, Nn);
        k_node_gemm2<<<ngrid, 256, 0, stream>>>(hidden, w_g2 + (size_t)l * hh,
                                                gin_b2 + (size_t)l * HDIM, x,
                                                (l < L - 1) ? 1 : 0, Nn);
    }

    // 7) output MLP + fused loss reduction
    k_zero<<<1, 256, 0, stream>>>(out, G);
    k_out_gemm1<<<(E + 127) / 128, 256, 0, stream>>>(x, row, col, edge_attr, w_o1, ob1,
                                                     h1s1, E);
    k_out_gemm2<<<(E + 127) / 128, 256, 0, stream>>>(h1s1, w_o2, ob2, ow3, ob3, z, eg,
                                                     out, E);
}
